// Brain_68521908240534
// MI455X (gfx1250) — matrix operand order repaired and checked by reading, not compiled, not run on hardware
//
#include <hip/hip_runtime.h>
#include <hip/hip_bf16.h>
#include <math.h>

// ---------------------------------------------------------------------------
// Brain RNN: r_{t+1} = amp * sigmoid(steep * (x_t*W_in + r_t @ W_rec^T + b - off))
// out = r_T @ out_w^T + out_b
// B=256, N=K=4096, T=100.  GEMM per step via v_wmma_f32_16x16x32_bf16.
// Single-wave workgroups -> 512 WGs/step to fill the MI455X.
// ---------------------------------------------------------------------------

typedef __attribute__((ext_vector_type(16))) __bf16 v16bf;
typedef __attribute__((ext_vector_type(4)))  __bf16 v4bf;
typedef __attribute__((ext_vector_type(8)))  float  v8f;

#define NN 4096   // neurons (N == K)
#define BB 256    // batch (M)
#define TSTEPS 100

union Frag16 {            // 16 bf16 per lane = 8 VGPRs (A or B operand of 16x16x32)
    v16bf v;
    int4  q[2];           // two 16-byte chunks
};

// --- convert W_rec fp32 -> bf16 (one pass, 16M elements, 4 per thread) -----
__global__ __launch_bounds__(256) void cvt_w_kernel(const float* __restrict__ W,
                                                    __bf16* __restrict__ Wb) {
    size_t i = ((size_t)blockIdx.x * 256 + threadIdx.x) * 4;
    float4 f = *(const float4*)(W + i);
    v4bf o = { (__bf16)f.x, (__bf16)f.y, (__bf16)f.z, (__bf16)f.w };
    *(v4bf*)(Wb + i) = o;
}

// --- zero the initial r buffer (bf16, 16 bytes per thread) -----------------
__global__ __launch_bounds__(256) void zero_kernel(int4* __restrict__ p) {
    p[(size_t)blockIdx.x * 256 + threadIdx.x] = int4{0, 0, 0, 0};
}

// --- one pondering step: fused GEMM (bf16 WMMA, f32 acc) + sigmoid ---------
// One wave per workgroup.  Wave tile: 64(M) x 32(N) = 4x2 16x16 accumulators.
// Grid: (4096/32, 256/64) = (128, 4) = 512 workgroups.
__global__ __launch_bounds__(32) void step_kernel(
    const __bf16* __restrict__ Wbf,   // [N][K] bf16 (row n contiguous along k)
    const __bf16* __restrict__ Rin,   // [M][K] bf16
    __bf16*       __restrict__ Rout,  // [M][N] bf16
    const float*  __restrict__ x,     // [B][T]
    int t,
    const float*  __restrict__ Winp,  // [N]
    const float*  __restrict__ bias,  // [N]
    const float*  __restrict__ steep, // [N]
    const float*  __restrict__ offs,  // [N]
    const float*  __restrict__ amp)   // [N]
{
    const int lane = threadIdx.x;        // 0..31 (wave32)
    const int lh   = lane >> 4;          // half-wave select
    const int l15  = lane & 15;

    const int m0 = blockIdx.y * 64;      // wave M base (4 x 16 tiles)
    const int n0 = blockIdx.x * 32;      // wave N base (2 x 16 tiles)

    v8f acc[4][2];
    #pragma unroll
    for (int tt = 0; tt < 4; ++tt)
        #pragma unroll
        for (int nt = 0; nt < 2; ++nt)
            acc[tt][nt] = (v8f)(0.0f);

    // A frag (16-bit 16x32 layout): lane-lo -> K{0..7,16..23}, lane-hi -> K{8..15,24..31}
    const __bf16* aptr[4];
    #pragma unroll
    for (int tt = 0; tt < 4; ++tt)
        aptr[tt] = Rin + (size_t)(m0 + tt * 16 + l15) * NN + (lh ? 8 : 0);

    // B frag (16-bit 32x16 layout): lane n = column n = W_rec row n; K contiguous.
    // lane-lo -> K{0..7,16..23}, lane-hi -> K{8..15,24..31}.
    const __bf16* bptr[2];
    #pragma unroll
    for (int nt = 0; nt < 2; ++nt)
        bptr[nt] = Wbf + (size_t)(n0 + nt * 16 + l15) * NN + (lh ? 8 : 0);

    for (int k = 0; k < NN; k += 32) {
        Frag16 a[4], b[2];
        #pragma unroll
        for (int tt = 0; tt < 4; ++tt) {
            a[tt].q[0] = *(const int4*)(aptr[tt] + k);
            a[tt].q[1] = *(const int4*)(aptr[tt] + k + 16);
        }
        #pragma unroll
        for (int nt = 0; nt < 2; ++nt) {
            b[nt].q[0] = *(const int4*)(bptr[nt] + k);
            b[nt].q[1] = *(const int4*)(bptr[nt] + k + 16);
        }
        #pragma unroll
        for (int tt = 0; tt < 4; ++tt)
            #pragma unroll
            for (int nt = 0; nt < 2; ++nt)
                acc[tt][nt] = __builtin_amdgcn_wmma_f32_16x16x32_bf16(
                    /*neg_a=*/false, a[tt].v,
                    /*neg_b=*/false, b[nt].v,
                    /*c_mod=*/(short)0, acc[tt][nt],
                    /*reuse_a=*/false, /*reuse_b=*/false);
    }

    // Epilogue. C/D layout: VGPR j -> (M = j + 8*lh, N = l15) within each 16x16 tile.
    float wi[2], bi[2], st[2], of[2], am[2];
    int   ncol[2];
    #pragma unroll
    for (int nt = 0; nt < 2; ++nt) {
        int n = n0 + nt * 16 + l15;
        ncol[nt] = n;
        wi[nt] = Winp[n];  bi[nt] = bias[n];
        st[nt] = steep[n]; of[nt] = offs[n]; am[nt] = amp[n];
    }
    #pragma unroll
    for (int tt = 0; tt < 4; ++tt) {
        const int mrow = m0 + tt * 16 + lh * 8;
        #pragma unroll
        for (int j = 0; j < 8; ++j) {
            const int m = mrow + j;
            const float xv = x[(size_t)m * TSTEPS + t];
            #pragma unroll
            for (int nt = 0; nt < 2; ++nt) {
                float lin = acc[tt][nt][j] + xv * wi[nt] + bi[nt];
                float r   = am[nt] / (1.0f + __expf(-st[nt] * (lin - of[nt])));
                Rout[(size_t)m * NN + ncol[nt]] = (__bf16)r;
            }
        }
    }
}

// --- final projection: out[b] = sum_n r[b][n]*out_w[n] + out_b -------------
__global__ __launch_bounds__(256) void out_kernel(const __bf16* __restrict__ R,
                                                  const float* __restrict__ ow,
                                                  const float* __restrict__ ob,
                                                  float* __restrict__ out) {
    __shared__ float red[256];
    const int b   = blockIdx.x;
    const int tid = threadIdx.x;
    float s = 0.0f;
    for (int n = tid; n < NN; n += 256)
        s += (float)R[(size_t)b * NN + n] * ow[n];
    red[tid] = s;
    __syncthreads();
    #pragma unroll
    for (int w = 128; w > 0; w >>= 1) {
        if (tid < w) red[tid] += red[tid + w];
        __syncthreads();
    }
    if (tid == 0) out[b] = red[0] + ob[0];
}

extern "C" void kernel_launch(void* const* d_in, const int* in_sizes, int n_in,
                              void* d_out, int out_size, void* d_ws, size_t ws_size,
                              hipStream_t stream) {
    const float* x     = (const float*)d_in[0];  // [256,100,1]
    const float* Winp  = (const float*)d_in[1];  // [4096,1]
    const float* Wrec  = (const float*)d_in[2];  // [4096,4096]
    const float* bias  = (const float*)d_in[3];  // [4096]
    const float* steep = (const float*)d_in[4];
    const float* offs  = (const float*)d_in[5];
    const float* amp   = (const float*)d_in[6];
    const float* ow    = (const float*)d_in[7];  // [1,4096]
    const float* ob    = (const float*)d_in[8];  // [1]
    float* out = (float*)d_out;                  // [256,1]

    char* ws = (char*)d_ws;
    __bf16* Wbf = (__bf16*)ws;                                           // 32 MB
    __bf16* r0  = (__bf16*)(ws + (size_t)NN * NN * sizeof(__bf16));      // 2 MB
    __bf16* r1  = r0 + (size_t)BB * NN;                                  // 2 MB

    // 1) W_rec fp32 -> bf16 (L2-resident thereafter: 32 MB << 192 MB L2)
    cvt_w_kernel<<<(NN * (size_t)NN) / (256 * 4), 256, 0, stream>>>(Wrec, Wbf);

    // 2) r_0 = 0
    zero_kernel<<<(BB * NN * sizeof(__bf16)) / (16 * 256), 256, 0, stream>>>((int4*)r0);

    // 3) 100 pondering steps (serial dependency -> one fused kernel each).
    //    512 single-wave workgroups per step for full-chip occupancy.
    dim3 grid(NN / 32, BB / 64);   // (128, 4)
    __bf16* cur = r0;
    __bf16* nxt = r1;
    for (int t = 0; t < TSTEPS; ++t) {
        step_kernel<<<grid, 32, 0, stream>>>(Wbf, cur, nxt, x, t,
                                             Winp, bias, steep, offs, amp);
        __bf16* tmp = cur; cur = nxt; nxt = tmp;
    }

    // 4) output projection
    out_kernel<<<BB, 256, 0, stream>>>(cur, ow, ob, out);
}
